// LocalTransformerPretrain_1400159338535
// MI455X (gfx1250) — matrix operand order repaired and checked by reading, not compiled, not run on hardware
//
#include <hip/hip_runtime.h>
#include <hip/hip_bf16.h>

// Problem sizes (fixed by the reference)
#define NN_ 4096
#define DD_ 1024
#define FF_ 4096
#define MM_ 2048
#define KK_ 307

// LDS tile strides (elements), padded to dodge bank conflicts, 16B aligned
#define SAS 40    // A tile row stride (32 data + 8 pad)
#define SBS 136   // B tile row stride (128 data + 8 pad)

typedef __attribute__((ext_vector_type(16))) __bf16 v16bf;
typedef __attribute__((ext_vector_type(8)))  float  v8f;

__device__ __forceinline__ unsigned short f2bf(float f) {
    unsigned u = __float_as_uint(f);
    u += 0x7FFFu + ((u >> 16) & 1u);   // round-to-nearest-even
    return (unsigned short)(u >> 16);
}

// Low 32 bits of a generic pointer to LDS = byte offset within the
// workgroup's LDS allocation (flat-address LDS aperture truncation).
__device__ __forceinline__ unsigned lds_off(const void* p) {
    return (unsigned)(unsigned long long)(uintptr_t)p;
}

// Async copy 16B global -> LDS (GLOBAL_LOAD_ASYNC_TO_LDS_B128, ASYNCcnt).
__device__ __forceinline__ void async_cp_b128(unsigned lds_byte, const void* gsrc) {
    asm volatile("global_load_async_to_lds_b128 %0, %1, off"
                 :: "v"(lds_byte), "v"((unsigned long long)gsrc)
                 : "memory");
}

__device__ __forceinline__ void wait_async0()   { asm volatile("s_wait_asynccnt 0x0" ::: "memory"); }
__device__ __forceinline__ void wait_async_le3(){ asm volatile("s_wait_asynccnt 0x3" ::: "memory"); }
__device__ __forceinline__ void wait_async_le6(){ asm volatile("s_wait_asynccnt 0x6" ::: "memory"); }

// ---------------------------------------------------------------------------
// bf16 WMMA GEMM: 4-deep async->LDS pipeline (stage distance 3, relaxed
// s_wait_asynccnt 0x3) + register-level fragment double buffering so the
// 4 WMMAs per slice issue back-to-back while the next slice's ds_loads and
// the slice+3 async transfers are in flight.
//   C[Mr,Nc] = A[Mr,Kd] @ B[Kd,Nc]   (row-major bf16 in, f32 accum)
// Block: 256 threads = 8 waves; block tile 64(M) x 128(N); k-step 32.
// ---------------------------------------------------------------------------
__device__ __forceinline__ void epilogue_tile(
    v8f c, int n, int mBase, int Nc,
    const float* bias, float scale, const unsigned char* mask, int do_gelu,
    float* Cf, unsigned short* C16, unsigned short* C16T, int ldt)
{
    float bv = bias ? bias[n] : 0.0f;
    unsigned short pk[8];
#pragma unroll
    for (int r = 0; r < 8; ++r) {
        float x = c[r] * scale + bv;
        if (do_gelu) x = 0.5f * x * (1.0f + erff(x * 0.70710678118654752f));
        if (mask && mask[(size_t)(mBase + r) * (size_t)Nc + n]) x = -1e9f;
        if (Cf)  Cf [(size_t)(mBase + r) * (size_t)Nc + n] = x;
        if (C16) C16[(size_t)(mBase + r) * (size_t)Nc + n] = f2bf(x);
        pk[r] = f2bf(x);
    }
    if (C16T) { // lane holds 8 consecutive M for one N -> contiguous b128 store
        int4 pv; __builtin_memcpy(&pv, pk, 16);
        *((int4*)(C16T + (size_t)n * (size_t)ldt + mBase)) = pv;
    }
}

#define WMMA_BF16(acc, a, b) \
    acc = __builtin_amdgcn_wmma_f32_16x16x32_bf16(false, a, false, b, (short)0, acc, false, false)

__global__ __launch_bounds__(256) void gemm_bf16_wmma(
    const unsigned short* __restrict__ A, const unsigned short* __restrict__ B,
    int Mr, int Nc, int Kd,
    const float* __restrict__ bias, float scale,
    const unsigned char* __restrict__ mask, int do_gelu,
    float* __restrict__ Cf, unsigned short* __restrict__ C16,
    unsigned short* __restrict__ C16T, int ldt)
{
    __shared__ unsigned short shA[4][64 * SAS];   // 4 x 5120 B
    __shared__ unsigned short shB[4][32 * SBS];   // 4 x 8704 B

    const int tid  = threadIdx.x;
    const int lane = tid & 31;
    const int wave = tid >> 5;
    const int m0   = blockIdx.y * 64;
    const int n0   = blockIdx.x * 128;

    // ---- cooperative staging addresses (per thread: 1 A chunk + 2 B chunks)
    const unsigned short* gA = A + (size_t)(m0 + (tid >> 2)) * Kd + (tid & 3) * 8;
    const unsigned aL = (unsigned)(((tid >> 2) * SAS + (tid & 3) * 8) * 2);
    const int bRow0 = tid >> 4;           // 0..15
    const int bCol  = (tid & 15) * 8;     // element offset
    const unsigned short* gB0 = B + (size_t)bRow0 * Nc + n0 + bCol;
    const unsigned short* gB1 = gB0 + (size_t)16 * Nc;   // rows 16..31
    const unsigned bL0 = (unsigned)((bRow0 * SBS + bCol) * 2);
    const unsigned bL1 = (unsigned)(((bRow0 + 16) * SBS + bCol) * 2);

    auto stage = [&](int buf, int k0) {
        async_cp_b128(lds_off(&shA[buf][0]) + aL,  gA  + k0);
        async_cp_b128(lds_off(&shB[buf][0]) + bL0, gB0 + (size_t)k0 * Nc);
        async_cp_b128(lds_off(&shB[buf][0]) + bL1, gB1 + (size_t)k0 * Nc);
    };

    // ---- fragment read addresses (within LDS tile)
    const int wm = wave >> 1, wn = wave & 1;
    const int aOffL = (wm * 16 + (lane & 15)) * SAS + ((lane >> 4) << 3);
    const int bOffL = ((lane >> 4) << 3) * SBS + wn * 64 + (lane & 15);

    auto ldA = [&](v16bf& a, const unsigned short* base) {
        const unsigned short* p = base + aOffL;
        ((int4*)&a)[0] = *(const int4*)(p);
        ((int4*)&a)[1] = *(const int4*)(p + 16);
    };
    auto ldB = [&](v16bf& b, const unsigned short* base, int off) {
        const unsigned short* p = base + bOffL + off;
        unsigned short t[16];
#pragma unroll
        for (int j = 0; j < 8; ++j) { t[j] = p[j * SBS]; t[8 + j] = p[(16 + j) * SBS]; }
        __builtin_memcpy(&b, t, 32);
    };

    v8f c0 = {}, c1 = {}, c2 = {}, c3 = {};
    v16bf aE, bE0, bE1, bE2, bE3;   // fragments for even slices
    v16bf aO, bO0, bO1, bO2, bO3;   // fragments for odd slices
    const int nIter = Kd / 32;      // multiple of 4 (>= 32)

    // ---- prologue: slices 0,1,2 staged; fragments of slice 0 in E
    stage(0, 0); stage(1, 32); stage(2, 64);
    wait_async_le6();     // slice 0 landed (1,2 may be in flight)
    __syncthreads();
    ldA(aE, &shA[0][0]);
    ldB(bE0, &shB[0][0], 0);  ldB(bE1, &shB[0][0], 16);
    ldB(bE2, &shB[0][0], 32); ldB(bE3, &shB[0][0], 48);

    for (int i = 0; i < nIter; i += 4) {
        // q0: compute slice i (E, buf0); frags i+1 from buf1; stage i+3 -> buf3
        wait_async_le3();                 // slice i+1 landed (i+2 in flight)
        __syncthreads();                  // + all waves drained prior LDS reads
        if (i + 3 < nIter) stage(3, (i + 3) * 32);
        ldA(aO, &shA[1][0]);
        ldB(bO0, &shB[1][0], 0);  ldB(bO1, &shB[1][0], 16);
        ldB(bO2, &shB[1][0], 32); ldB(bO3, &shB[1][0], 48);
        WMMA_BF16(c0, aE, bE0); WMMA_BF16(c1, aE, bE1);
        WMMA_BF16(c2, aE, bE2); WMMA_BF16(c3, aE, bE3);

        // q1: compute slice i+1 (O); frags i+2 from buf2; stage i+4 -> buf0
        wait_async_le3();                 // slice i+2 landed (i+3 in flight)
        __syncthreads();
        if (i + 4 < nIter) stage(0, (i + 4) * 32);
        ldA(aE, &shA[2][0]);
        ldB(bE0, &shB[2][0], 0);  ldB(bE1, &shB[2][0], 16);
        ldB(bE2, &shB[2][0], 32); ldB(bE3, &shB[2][0], 48);
        WMMA_BF16(c0, aO, bO0); WMMA_BF16(c1, aO, bO1);
        WMMA_BF16(c2, aO, bO2); WMMA_BF16(c3, aO, bO3);

        // q2: compute slice i+2 (E); frags i+3 from buf3; stage i+5 -> buf1
        if (i + 4 < nIter) wait_async_le3(); else wait_async0(); // need i+3 done
        __syncthreads();
        if (i + 5 < nIter) stage(1, (i + 5) * 32);
        ldA(aO, &shA[3][0]);
        ldB(bO0, &shB[3][0], 0);  ldB(bO1, &shB[3][0], 16);
        ldB(bO2, &shB[3][0], 32); ldB(bO3, &shB[3][0], 48);
        WMMA_BF16(c0, aE, bE0); WMMA_BF16(c1, aE, bE1);
        WMMA_BF16(c2, aE, bE2); WMMA_BF16(c3, aE, bE3);

        // q3: compute slice i+3 (O); frags i+4 from buf0; stage i+6 -> buf2
        if (i + 5 < nIter) wait_async_le3(); else wait_async0(); // need i+4 done
        __syncthreads();
        if (i + 6 < nIter) stage(2, (i + 6) * 32);
        if (i + 4 < nIter) {
            ldA(aE, &shA[0][0]);
            ldB(bE0, &shB[0][0], 0);  ldB(bE1, &shB[0][0], 16);
            ldB(bE2, &shB[0][0], 32); ldB(bE3, &shB[0][0], 48);
        }
        WMMA_BF16(c0, aO, bO0); WMMA_BF16(c1, aO, bO1);
        WMMA_BF16(c2, aO, bO2); WMMA_BF16(c3, aO, bO3);
    }

    const int tileM = m0 + wm * 16;
    const int nCol0 = n0 + wn * 64 + (lane & 15);
    const int mBase = tileM + ((lane >> 4) << 3);
    epilogue_tile(c0, nCol0 +  0, mBase, Nc, bias, scale, mask, do_gelu, Cf, C16, C16T, ldt);
    epilogue_tile(c1, nCol0 + 16, mBase, Nc, bias, scale, mask, do_gelu, Cf, C16, C16T, ldt);
    epilogue_tile(c2, nCol0 + 32, mBase, Nc, bias, scale, mask, do_gelu, Cf, C16, C16T, ldt);
    epilogue_tile(c3, nCol0 + 48, mBase, Nc, bias, scale, mask, do_gelu, Cf, C16, C16T, ldt);
}

// ---------------------------------------------------------------------------
// Row softmax (one block per row). Of may alias S (in-place). Writes bf16 too.
// ---------------------------------------------------------------------------
__global__ __launch_bounds__(256) void softmax_row_kernel(
    const float* __restrict__ S, float* __restrict__ Of,
    unsigned short* __restrict__ Ob, int Nc)
{
    __shared__ float red[256];
    const int row = blockIdx.x;
    const float* s = S + (size_t)row * Nc;
    float m = -3.4e38f;
    for (int i = threadIdx.x; i < Nc; i += 256) m = fmaxf(m, s[i]);
    red[threadIdx.x] = m; __syncthreads();
    for (int w = 128; w > 0; w >>= 1) {
        if (threadIdx.x < w) red[threadIdx.x] = fmaxf(red[threadIdx.x], red[threadIdx.x + w]);
        __syncthreads();
    }
    m = red[0]; __syncthreads();
    float sum = 0.0f;
    for (int i = threadIdx.x; i < Nc; i += 256) sum += expf(s[i] - m);
    red[threadIdx.x] = sum; __syncthreads();
    for (int w = 128; w > 0; w >>= 1) {
        if (threadIdx.x < w) red[threadIdx.x] += red[threadIdx.x + w];
        __syncthreads();
    }
    const float inv = 1.0f / red[0];
    for (int i = threadIdx.x; i < Nc; i += 256) {
        float a = expf(s[i] - m) * inv;
        if (Of) Of[(size_t)row * Nc + i] = a;
        Ob[(size_t)row * Nc + i] = f2bf(a);
    }
}

// ---------------------------------------------------------------------------
// out = LayerNorm(A + R) * g + b  (one block per row, D=1024). Of may alias R.
// ---------------------------------------------------------------------------
__global__ __launch_bounds__(256) void ln_residual_kernel(
    const float* __restrict__ A, const float* __restrict__ R,
    const float* __restrict__ g, const float* __restrict__ b,
    float* __restrict__ Of, unsigned short* __restrict__ Ob)
{
    __shared__ float red[256];
    __shared__ float red2[256];
    const int row = blockIdx.x;
    const float* pa = A + (size_t)row * DD_;
    const float* pr = R + (size_t)row * DD_;
    float v[4];
    float s = 0.0f, sq = 0.0f;
#pragma unroll
    for (int j = 0; j < 4; ++j) {
        int i = threadIdx.x + j * 256;
        float x = pa[i] + pr[i];
        v[j] = x; s += x; sq += x * x;
    }
    red[threadIdx.x] = s; red2[threadIdx.x] = sq; __syncthreads();
    for (int w = 128; w > 0; w >>= 1) {
        if (threadIdx.x < w) {
            red[threadIdx.x]  += red[threadIdx.x + w];
            red2[threadIdx.x] += red2[threadIdx.x + w];
        }
        __syncthreads();
    }
    const float mu  = red[0] * (1.0f / DD_);
    const float var = red2[0] * (1.0f / DD_) - mu * mu;
    const float rs  = rsqrtf(var + 1e-5f);
#pragma unroll
    for (int j = 0; j < 4; ++j) {
        int i = threadIdx.x + j * 256;
        float y = (v[j] - mu) * rs * g[i] + b[i];
        Of[(size_t)row * DD_ + i] = y;
        Ob[(size_t)row * DD_ + i] = f2bf(y);
    }
}

// --------------------------- small element kernels -------------------------
__global__ void copy_cvt_kernel(const float* __restrict__ x,
                                float* __restrict__ Uf,
                                unsigned short* __restrict__ Ub, size_t n)
{
    for (size_t i = (size_t)blockIdx.x * 256 + threadIdx.x; i < n;
         i += (size_t)gridDim.x * 256) {
        float v = x[i];
        if (Uf) Uf[i] = v;
        Ub[i] = f2bf(v);
    }
}

__global__ void mask_scatter_kernel(const float* __restrict__ x,
                                    const int* __restrict__ mask_nodes,
                                    const int* __restrict__ gene_idx,
                                    float* __restrict__ Uf,
                                    unsigned short* __restrict__ Ub,
                                    float* __restrict__ x_init)
{
    int id = blockIdx.x * 256 + threadIdx.x;
    if (id >= MM_ * KK_) return;
    int m = id / KK_;
    int r = mask_nodes[m];
    int c = gene_idx[id];
    x_init[id] = x[(size_t)r * DD_ + c];
    Uf[(size_t)r * DD_ + c] = 0.0f;
    Ub[(size_t)r * DD_ + c] = 0;
}

__global__ void gather_rows_bf16_kernel(const float* __restrict__ recon,
                                        const int* __restrict__ mask_nodes,
                                        unsigned short* __restrict__ outb)
{
    int id = blockIdx.x * 256 + threadIdx.x;
    if (id >= MM_ * DD_) return;
    int m = id / DD_, d = id % DD_;
    outb[id] = f2bf(recon[(size_t)mask_nodes[m] * DD_ + d]);
}

__global__ void gather_xrecon_kernel(const float* __restrict__ hM,
                                     const int* __restrict__ gene_idx,
                                     float* __restrict__ xr)
{
    int id = blockIdx.x * 256 + threadIdx.x;
    if (id >= MM_ * KK_) return;
    int m = id / KK_;
    xr[id] = hM[(size_t)m * DD_ + gene_idx[id]];
}

// ---------------------------------------------------------------------------
extern "C" void kernel_launch(void* const* d_in, const int* in_sizes, int n_in,
                              void* d_out, int out_size, void* d_ws, size_t ws_size,
                              hipStream_t stream)
{
    (void)in_sizes; (void)n_in; (void)out_size; (void)ws_size;
    const float* x               = (const float*)d_in[0];
    const unsigned char* rmask   = (const unsigned char*)d_in[1]; // bool [N,N]
    const int* mask_nodes        = (const int*)d_in[3];
    const int* gene_idx          = (const int*)d_in[4];
    const float* wq1 = (const float*)d_in[5];
    const float* wk1 = (const float*)d_in[6];
    const float* wv1 = (const float*)d_in[7];
    const float* g1  = (const float*)d_in[8];
    const float* be1 = (const float*)d_in[9];
    const float* fw1 = (const float*)d_in[10];
    const float* fb1 = (const float*)d_in[11];
    const float* fw2 = (const float*)d_in[12];
    const float* fb2 = (const float*)d_in[13];
    const float* g2  = (const float*)d_in[14];
    const float* be2 = (const float*)d_in[15];
    const float* wq2 = (const float*)d_in[16];
    const float* wk2 = (const float*)d_in[17];
    const float* wv2 = (const float*)d_in[18];
    const float* g3  = (const float*)d_in[19];
    const float* be3 = (const float*)d_in[20];
    const float* hw1 = (const float*)d_in[21];
    const float* hb1 = (const float*)d_in[22];
    const float* hw2 = (const float*)d_in[23];
    const float* hb2 = (const float*)d_in[24];
    const float* g4  = (const float*)d_in[25];
    const float* be4 = (const float*)d_in[26];
    const float* head_w = (const float*)d_in[27];
    const float* head_b = (const float*)d_in[28];

    const size_t ND = (size_t)NN_ * DD_;
    const size_t NNm = (size_t)NN_ * NN_;
    const size_t NF = (size_t)NN_ * FF_;
    const size_t DF = (size_t)DD_ * FF_;
    const size_t DDsq = (size_t)DD_ * DD_;
    const size_t MK = (size_t)MM_ * KK_;
    const size_t MD = (size_t)MM_ * DD_;

    // Output layout (flat f32): x_init, x_recon, encode_weights, recon
    float* out      = (float*)d_out;
    float* o_xinit  = out;
    float* o_xrec   = out + MK;
    float* o_enc    = out + 2 * MK;
    float* o_recon  = out + 2 * MK + NNm;

    // Workspace bump allocator
    char* wsp = (char*)d_ws;
    size_t off = 0;
    auto alloc = [&](size_t bytes) -> void* {
        void* p = (void*)(wsp + off);
        off = (off + bytes + 255) & ~(size_t)255;
        return p;
    };
    float*          Uf   = (float*)alloc(ND * 4);           // use_x f32
    unsigned short* X16  = (unsigned short*)alloc(ND * 2);  // current activation bf16
    unsigned short* q16  = (unsigned short*)alloc(ND * 2);
    unsigned short* kT16 = (unsigned short*)alloc(ND * 2);  // [D,N]
    unsigned short* v16  = (unsigned short*)alloc(ND * 2);
    float*          S    = (float*)alloc(NNm * 4);          // scores
    unsigned short* A16  = (unsigned short*)alloc(NNm * 2); // softmax bf16
    float*          T0   = (float*)alloc(ND * 4);           // gemm out
    float*          H    = (float*)alloc(ND * 4);           // hidden f32
    unsigned short* t16  = (unsigned short*)alloc(NF * 2);  // ffn mid bf16
    unsigned short* wq1b = (unsigned short*)alloc(DDsq * 2);
    unsigned short* wk1b = (unsigned short*)alloc(DDsq * 2);
    unsigned short* wv1b = (unsigned short*)alloc(DDsq * 2);
    unsigned short* wq2b = (unsigned short*)alloc(DDsq * 2);
    unsigned short* wk2b = (unsigned short*)alloc(DDsq * 2);
    unsigned short* wv2b = (unsigned short*)alloc(DDsq * 2);
    unsigned short* hwb  = (unsigned short*)alloc(DDsq * 2); // head_w
    unsigned short* fw1b = (unsigned short*)alloc(DF * 2);
    unsigned short* fw2b = (unsigned short*)alloc(DF * 2);
    unsigned short* hw1b = (unsigned short*)alloc(DF * 2);
    unsigned short* hw2b = (unsigned short*)alloc(DF * 2);
    unsigned short* rM16 = (unsigned short*)alloc(MD * 2);  // gathered recon rows
    float*          hM   = (float*)alloc(MD * 4);           // head out (M rows)

    auto cvt = [&](const float* src, unsigned short* dst, size_t n) {
        copy_cvt_kernel<<<dim3(4096), dim3(256), 0, stream>>>(src, nullptr, dst, n);
    };
    auto gemm = [&](const unsigned short* A, const unsigned short* B,
                    int Mr, int Nc, int Kd, const float* bias, float scale,
                    const unsigned char* mask, int gelu,
                    float* Cf, unsigned short* C16, unsigned short* C16T, int ldt) {
        dim3 grid((unsigned)(Nc / 128), (unsigned)(Mr / 64));
        gemm_bf16_wmma<<<grid, dim3(256), 0, stream>>>(
            A, B, Mr, Nc, Kd, bias, scale, mask, gelu, Cf, C16, C16T, ldt);
    };

    // Convert weights to bf16
    cvt(wq1, wq1b, DDsq); cvt(wk1, wk1b, DDsq); cvt(wv1, wv1b, DDsq);
    cvt(wq2, wq2b, DDsq); cvt(wk2, wk2b, DDsq); cvt(wv2, wv2b, DDsq);
    cvt(head_w, hwb, DDsq);
    cvt(fw1, fw1b, DF); cvt(fw2, fw2b, DF);
    cvt(hw1, hw1b, DF); cvt(hw2, hw2b, DF);

    // use_x = x with masked genes zeroed; x_init gather
    copy_cvt_kernel<<<dim3(4096), dim3(256), 0, stream>>>(x, Uf, X16, ND);
    mask_scatter_kernel<<<dim3((MM_ * KK_ + 255) / 256), dim3(256), 0, stream>>>(
        x, mask_nodes, gene_idx, Uf, X16, o_xinit);

    const float invsq = 0.03125f; // 1/sqrt(1024)

    // ---------------- Layer 1 (encoder) ----------------
    gemm(X16, wq1b, NN_, DD_, DD_, nullptr, 1.0f, nullptr, 0, nullptr, q16, nullptr, 0);
    gemm(X16, wk1b, NN_, DD_, DD_, nullptr, 1.0f, nullptr, 0, nullptr, nullptr, kT16, NN_);
    gemm(X16, wv1b, NN_, DD_, DD_, nullptr, 1.0f, nullptr, 0, nullptr, v16, nullptr, 0);
    gemm(q16, kT16, NN_, NN_, DD_, nullptr, invsq, rmask, 0, S, nullptr, nullptr, 0);
    softmax_row_kernel<<<dim3(NN_), dim3(256), 0, stream>>>(S, o_enc, A16, NN_); // encode_weights
    gemm(A16, v16, NN_, DD_, NN_, nullptr, 1.0f, nullptr, 0, T0, nullptr, nullptr, 0);
    ln_residual_kernel<<<dim3(NN_), dim3(256), 0, stream>>>(T0, Uf, g1, be1, H, X16);
    gemm(X16, fw1b, NN_, FF_, DD_, fb1, 1.0f, nullptr, 1, nullptr, t16, nullptr, 0);
    gemm(t16, fw2b, NN_, DD_, FF_, fb2, 1.0f, nullptr, 0, T0, nullptr, nullptr, 0);
    ln_residual_kernel<<<dim3(NN_), dim3(256), 0, stream>>>(T0, H, g2, be2, H, X16);

    // ---------------- Layer 2 (decoder) ----------------
    gemm(X16, wq2b, NN_, DD_, DD_, nullptr, 1.0f, nullptr, 0, nullptr, q16, nullptr, 0);
    gemm(X16, wk2b, NN_, DD_, DD_, nullptr, 1.0f, nullptr, 0, nullptr, nullptr, kT16, NN_);
    gemm(X16, wv2b, NN_, DD_, DD_, nullptr, 1.0f, nullptr, 0, nullptr, v16, nullptr, 0);
    gemm(q16, kT16, NN_, NN_, DD_, nullptr, invsq, rmask, 0, S, nullptr, nullptr, 0);
    softmax_row_kernel<<<dim3(NN_), dim3(256), 0, stream>>>(S, nullptr, A16, NN_);
    gemm(A16, v16, NN_, DD_, NN_, nullptr, 1.0f, nullptr, 0, T0, nullptr, nullptr, 0);
    ln_residual_kernel<<<dim3(NN_), dim3(256), 0, stream>>>(T0, H, g3, be3, H, X16);
    gemm(X16, hw1b, NN_, FF_, DD_, hb1, 1.0f, nullptr, 1, nullptr, t16, nullptr, 0);
    gemm(t16, hw2b, NN_, DD_, FF_, hb2, 1.0f, nullptr, 0, T0, nullptr, nullptr, 0);
    ln_residual_kernel<<<dim3(NN_), dim3(256), 0, stream>>>(T0, H, g4, be4, o_recon, X16);

    // ---------------- Head on masked rows only ----------------
    gather_rows_bf16_kernel<<<dim3((MM_ * DD_ + 255) / 256), dim3(256), 0, stream>>>(
        o_recon, mask_nodes, rM16);
    gemm(rM16, hwb, MM_, DD_, DD_, head_b, 1.0f, nullptr, 0, hM, nullptr, nullptr, 0);
    gather_xrecon_kernel<<<dim3((MM_ * KK_ + 255) / 256), dim3(256), 0, stream>>>(
        hM, gene_idx, o_xrec);
}
